// Net_36799279792943
// MI455X (gfx1250) — matrix operand order repaired and checked by reading, not compiled, not run on hardware
//
#include <hip/hip_runtime.h>
#include <hip/hip_bf16.h>
#include <math.h>

typedef __attribute__((ext_vector_type(16))) _Float16 v16h;
typedef __attribute__((ext_vector_type(8)))  float    v8f;

// ---------------------------------------------------------------------------
// Elementwise helpers
// ---------------------------------------------------------------------------
__global__ __launch_bounds__(256) void fill_f32(float* __restrict__ p, float v, long n) {
    long i = (long)blockIdx.x * blockDim.x + threadIdx.x;
    long stride = (long)gridDim.x * blockDim.x;
    for (; i < n; i += stride) p[i] = v;
}

__global__ __launch_bounds__(256) void deg_acc(const long long* __restrict__ dst,
                                               float* __restrict__ deg, int E) {
    long i = (long)blockIdx.x * blockDim.x + threadIdx.x;
    long stride = (long)gridDim.x * blockDim.x;
    for (; i < E; i += stride) atomicAdd(&deg[(int)dst[i]], 1.0f);
}

__global__ __launch_bounds__(256) void deg_to_dinv(float* __restrict__ deg, int N) {
    long i = (long)blockIdx.x * blockDim.x + threadIdx.x;
    long stride = (long)gridDim.x * blockDim.x;
    for (; i < N; i += stride) deg[i] = rsqrtf(deg[i]);  // deg >= 1 always (self-loop)
}

// ---------------------------------------------------------------------------
// WMMA GEMM:  C[M x NCREAL] = A[M x KDIM] * B[KDIM x NCREAL]
// f32 inputs converted to f16 on the fly, f32 accumulate via
// v_wmma_f32_16x16x32_f16.  B is pre-swizzled into LDS in exact B-fragment
// order so each lane's fragment is one contiguous 32B read.
//
// A fragment (16-bit A 16x32, same k order as B): lane<16 holds row M=lane,
//   halves 0..15 = K k0..k0+15;
//   lane>=16 holds row M=lane-16, K k0+16..k0+31.
// B fragment (32x16): lane n<16 -> col n, K k0..k0+15 (2 per VGPR);
//   lane n+16 -> col n, K k0+16..k0+31.
// C/D: VGPR j -> M = j (lanes 0-15) / M = 8+j (lanes 16-31), N = lane&15.
// ---------------------------------------------------------------------------
template <int KDIM, int NT, int NCREAL>
__global__ __launch_bounds__(256) void wmma_gemm(const float* __restrict__ A,
                                                 const float* __restrict__ B,
                                                 float* __restrict__ C, int M) {
    constexpr int NPAD = NT * 16;
    constexpr int KB   = KDIM / 32;
    __shared__ _Float16 bsh[KB][NT][32][16];  // [kblock][coltile][lane][half]

    const int tid = threadIdx.x;
    // Stage B (convert f32 -> f16, pad cols >= NCREAL with zero), swizzled to
    // fragment order: element (k, c): lane = (c&15) + 16*((k&31)>=16),
    // half = (k&31)&15.
    for (int idx = tid; idx < KDIM * NPAD; idx += 256) {
        int k = idx / NPAD, c = idx % NPAD;
        float val = (c < NCREAL) ? B[(size_t)k * NCREAL + c] : 0.0f;
        int kb = k >> 5, s = k & 31;
        int t = c >> 4, n = c & 15;
        bsh[kb][t][n + ((s >= 16) ? 16 : 0)][s & 15] = (_Float16)val;
    }
    __syncthreads();

    const int wave    = tid >> 5;
    const int lane    = tid & 31;
    const int rowBase = blockIdx.x * 128 + wave * 16;

    int r = rowBase + (lane & 15);
    if (r >= M) r = M - 1;  // clamp loads; stores predicated below
    const int koff = (lane >= 16) ? 16 : 0;

    v8f acc[NT];
#pragma unroll
    for (int t = 0; t < NT; ++t) acc[t] = v8f{};

    for (int kb = 0; kb < KB; ++kb) {
        const float* ap = A + (size_t)r * KDIM + kb * 32 + koff;
        v16h afrag;
#pragma unroll
        for (int i = 0; i < 8; ++i) afrag[i] = (_Float16)ap[i];
#pragma unroll
        for (int i = 0; i < 8; ++i) afrag[8 + i] = (_Float16)ap[8 + i];

#pragma unroll
        for (int t = 0; t < NT; ++t) {
            v16h bfrag = *(const v16h*)&bsh[kb][t][lane][0];
            acc[t] = __builtin_amdgcn_wmma_f32_16x16x32_f16(
                false, afrag, false, bfrag, (short)0, acc[t], false, false);
        }
    }

    const int rstore = rowBase + ((lane >= 16) ? 8 : 0);
    const int cbase  = lane & 15;
#pragma unroll
    for (int t = 0; t < NT; ++t) {
        int col = t * 16 + cbase;
        if (col < NCREAL) {
#pragma unroll
            for (int j = 0; j < 8; ++j) {
                int rr = rstore + j;
                if (rr < M) C[(size_t)rr * NCREAL + col] = acc[t][j];
            }
        }
    }
}

// ---------------------------------------------------------------------------
// Edge aggregation: acc[dst] += h[src] * dinv[src] * dinv[dst]
// F=128: one wave per edge, float4 per lane. F=40: one wave per edge, scalar.
// ---------------------------------------------------------------------------
__global__ __launch_bounds__(256) void scatter_f128(const long long* __restrict__ src,
                                                    const long long* __restrict__ dst,
                                                    const float* __restrict__ dinv,
                                                    const float* __restrict__ h,
                                                    float* __restrict__ acc, int E) {
    long gid   = (long)blockIdx.x * blockDim.x + threadIdx.x;
    int  lane  = (int)(gid & 31);
    long wave  = gid >> 5;
    long nwave = ((long)gridDim.x * blockDim.x) >> 5;
    for (long e = wave; e < E; e += nwave) {
        int s = (int)src[e], d = (int)dst[e];
        float norm = dinv[s] * dinv[d];
        float4 m = ((const float4*)(h + (size_t)s * 128))[lane];
        float* p = acc + (size_t)d * 128 + lane * 4;
        atomicAdd(p + 0, m.x * norm);
        atomicAdd(p + 1, m.y * norm);
        atomicAdd(p + 2, m.z * norm);
        atomicAdd(p + 3, m.w * norm);
    }
}

__global__ __launch_bounds__(256) void scatter_f40(const long long* __restrict__ src,
                                                   const long long* __restrict__ dst,
                                                   const float* __restrict__ dinv,
                                                   const float* __restrict__ h,
                                                   float* __restrict__ acc, int E) {
    long gid   = (long)blockIdx.x * blockDim.x + threadIdx.x;
    int  lane  = (int)(gid & 31);
    long wave  = gid >> 5;
    long nwave = ((long)gridDim.x * blockDim.x) >> 5;
    for (long e = wave; e < E; e += nwave) {
        int s = (int)src[e], d = (int)dst[e];
        float norm = dinv[s] * dinv[d];
        for (int f = lane; f < 40; f += 32)
            atomicAdd(acc + (size_t)d * 40 + f, h[(size_t)s * 40 + f] * norm);
    }
}

// acc = relu(acc + h1 * dinv[n]^2 + b1), in place, F=128, float4 vectorized.
__global__ __launch_bounds__(256) void finalize_relu128(float* __restrict__ acc,
                                                        const float* __restrict__ h1,
                                                        const float* __restrict__ dinv,
                                                        const float* __restrict__ b1, int N) {
    long i = (long)blockIdx.x * blockDim.x + threadIdx.x;
    long stride = (long)gridDim.x * blockDim.x;
    long total = (long)N * 32;  // float4 chunks
    for (; i < total; i += stride) {
        int n = (int)(i >> 5), f4 = (int)(i & 31);
        float dn = dinv[n], s = dn * dn;
        float4 a = ((const float4*)acc)[i];
        float4 h = ((const float4*)h1)[i];
        float4 b = ((const float4*)b1)[f4];
        float4 r;
        r.x = fmaxf(fmaf(h.x, s, a.x) + b.x, 0.0f);
        r.y = fmaxf(fmaf(h.y, s, a.y) + b.y, 0.0f);
        r.z = fmaxf(fmaf(h.z, s, a.z) + b.z, 0.0f);
        r.w = fmaxf(fmaf(h.w, s, a.w) + b.w, 0.0f);
        ((float4*)acc)[i] = r;
    }
}

// out row += self-loop + bias, then log_softmax over 40 classes (in place).
__global__ __launch_bounds__(256) void finalize_logsoftmax40(float* __restrict__ out,
                                                             const float* __restrict__ h2,
                                                             const float* __restrict__ dinv,
                                                             const float* __restrict__ b2, int N) {
    long n = (long)blockIdx.x * blockDim.x + threadIdx.x;
    long stride = (long)gridDim.x * blockDim.x;
    for (; n < N; n += stride) {
        float dn = dinv[n], s = dn * dn;
        float v[40];
        float m = -3.402823466e+38f;
        for (int c = 0; c < 40; ++c) {
            float x = out[n * 40 + c] + h2[n * 40 + c] * s + b2[c];
            v[c] = x;
            m = fmaxf(m, x);
        }
        float sum = 0.0f;
        for (int c = 0; c < 40; ++c) sum += __expf(v[c] - m);
        float lse = m + __logf(sum);
        for (int c = 0; c < 40; ++c) out[n * 40 + c] = v[c] - lse;
    }
}

// ---------------------------------------------------------------------------
static inline size_t align256(size_t x) { return (x + 255) & ~(size_t)255; }

extern "C" void kernel_launch(void* const* d_in, const int* in_sizes, int n_in,
                              void* d_out, int out_size, void* d_ws, size_t ws_size,
                              hipStream_t stream) {
    const float*     x   = (const float*)d_in[0];      // [N, 256]
    const long long* ei  = (const long long*)d_in[1];  // [2, E] int64
    const float*     W1  = (const float*)d_in[2];      // [256, 128]
    const float*     b1  = (const float*)d_in[3];      // [128]
    const float*     W2  = (const float*)d_in[4];      // [128, 40]
    const float*     b2  = (const float*)d_in[5];      // [40]
    float*           out = (float*)d_out;              // [N, 40]

    const int H = in_sizes[3];           // 128
    const int F = in_sizes[2] / H;       // 256
    const int N = in_sizes[0] / F;       // 100000
    const int E = in_sizes[1] / 2;       // 1600000
    (void)n_in; (void)out_size; (void)ws_size;

    const long long* src = ei;
    const long long* dst = ei + E;

    // Workspace carve-up (~119 MB)
    char* ws = (char*)d_ws;
    size_t off = 0;
    float* dinv = (float*)(ws + off); off += align256((size_t)N * 4);
    float* h1   = (float*)(ws + off); off += align256((size_t)N * 128 * 4);
    float* acc1 = (float*)(ws + off); off += align256((size_t)N * 128 * 4);
    float* h2   = (float*)(ws + off); off += align256((size_t)N * 40 * 4);

    const int T = 256;
    // 1) degree (init 1.0 = self-loop), accumulate, rsqrt in place -> dinv
    fill_f32<<<(N + T - 1) / T, T, 0, stream>>>(dinv, 1.0f, N);
    deg_acc<<<(E + T - 1) / T, T, 0, stream>>>(dst, dinv, E);
    deg_to_dinv<<<(N + T - 1) / T, T, 0, stream>>>(dinv, N);

    // 2) h1 = x @ W1  (WMMA f16 -> f32)
    wmma_gemm<256, 8, 128><<<(N + 127) / 128, T, 0, stream>>>(x, W1, h1, N);

    // 3) acc1 = scatter(norm * h1[src]) ; then relu(acc1 + self + b1) in place
    fill_f32<<<(long)N * 128 / T, T, 0, stream>>>(acc1, 0.0f, (long)N * 128);
    scatter_f128<<<(int)(((long)E * 32 + T - 1) / T), T, 0, stream>>>(src, dst, dinv, h1, acc1, E);
    finalize_relu128<<<(int)(((long)N * 32 + T - 1) / T), T, 0, stream>>>(acc1, h1, dinv, b1, N);

    // 4) h2 = acc1 @ W2  (WMMA, 3 col tiles of 16, 40 real cols)
    wmma_gemm<128, 3, 40><<<(N + 127) / 128, T, 0, stream>>>(acc1, W2, h2, N);

    // 5) out = scatter(norm * h2[src]) ; finalize + log_softmax in place
    fill_f32<<<(int)(((long)N * 40 + T - 1) / T), T, 0, stream>>>(out, 0.0f, (long)N * 40);
    scatter_f40<<<(int)(((long)E * 32 + T - 1) / T), T, 0, stream>>>(src, dst, dinv, h2, out, E);
    finalize_logsoftmax40<<<(N + T - 1) / T, T, 0, stream>>>(out, h2, dinv, b2, N);
}
